// CrossCondGPT2_15212774162752
// MI455X (gfx1250) — matrix operand order repaired and checked by reading, not compiled, not run on hardware
//
#include <hip/hip_runtime.h>
#include <hip/hip_bf16.h>

// MI455X / gfx1250, wave32. All matrix math via v_wmma_f32_16x16x32_f16.

typedef __attribute__((ext_vector_type(16))) _Float16 v16h;
typedef __attribute__((ext_vector_type(8)))  _Float16 v8h;
typedef __attribute__((ext_vector_type(8)))  float    v8f;

#define C_DIM 768
#define NH    12
#define HD    64
#define T_MOT 512
#define L_SEQ 1616          // 3*512 + 5*16
#define LP    1632          // padded to multiple of 32 for the attention K loop
#define ML    (3*T_MOT)     // 1536
#define QT_PER (L_SEQ/16)   // 101 q-tiles per (b,h)

__device__ __forceinline__ v8f wmma_f16(v16h a, v16h b, v8f c) {
  // D = A(16x32 f16) * B(32x16 f16) + C(16x16 f32)
  return __builtin_amdgcn_wmma_f32_16x16x32_f16(false, a, false, b, (short)0, c,
                                                false, false);
}

// A-fragment (16x32 f16) from a row-major f16 matrix.
// Lane l holds row (l&15); halves 0-15 = K[k0 + (l>=16)*16 .. +15]
// (contiguous, matching load_bfrag).
__device__ __forceinline__ v16h load_afrag(const _Float16* __restrict__ base,
                                           int row, int stride, int k0, int lhi) {
  const _Float16* p = base + (size_t)row * stride + k0 + lhi * 16;
  union { v16h v; v8h h[2]; } u;
  u.h[0] = *(const v8h*)p;
  u.h[1] = *(const v8h*)(p + 8);
  return u.v;
}

// B-fragment (32x16 f16): lane l holds column (l&15), 16 contiguous K values
// starting at k0 + (l>=16)*16, read from a [N x K]-contiguous matrix.
__device__ __forceinline__ v16h load_bfrag(const _Float16* __restrict__ base,
                                           int col, int stride, int k0, int lhi) {
  return *(const v16h*)(base + (size_t)col * stride + k0 + lhi * 16);
}

__global__ void cvt_f32_f16(const float* __restrict__ in,
                            _Float16* __restrict__ out, int n) {
  int i = blockIdx.x * blockDim.x + threadIdx.x;
  if (i < n) out[i] = (_Float16)in[i];
}

// One wave per (32-row M pair, head-sized 64-col N group, {q,k,v}).
// 8 WMMAs per k-step on 2 A-frags x 4 B-frags (B reused across both M tiles).
__global__ void qkv_proj(const _Float16* __restrict__ x16,
                         const _Float16* __restrict__ Wq16, const float* __restrict__ bq,
                         const _Float16* __restrict__ Wk16, const float* __restrict__ bk,
                         const _Float16* __restrict__ Wv16, const float* __restrict__ bv,
                         _Float16* __restrict__ q16, _Float16* __restrict__ k16,
                         _Float16* __restrict__ vT16, int nwork) {
  int wid = blockIdx.x * (blockDim.x >> 5) + (threadIdx.x >> 5);
  if (wid >= nwork) return;
  int lane = threadIdx.x & 31;
  int lo = lane & 15, lhi = lane >> 4;
  int which = wid % 3;
  int rest  = wid / 3;
  int ng  = rest % NH;
  int mt2 = rest / NH;                           // pair of 16-row M tiles
  const _Float16* W   = (which == 0) ? Wq16 : (which == 1) ? Wk16 : Wv16;
  const float*   bias = (which == 0) ? bq   : (which == 1) ? bk   : bv;

  v8f zero = {0.f,0.f,0.f,0.f,0.f,0.f,0.f,0.f};
  v8f acc[2][4];
#pragma unroll
  for (int t = 0; t < 2; ++t)
#pragma unroll
    for (int s = 0; s < 4; ++s) acc[t][s] = zero;

  int arow0 = mt2 * 32 + lo;
  for (int k0 = 0; k0 < C_DIM; k0 += 32) {
    v16h a0 = load_afrag(x16, arow0,      C_DIM, k0, lhi);
    v16h a1 = load_afrag(x16, arow0 + 16, C_DIM, k0, lhi);
#pragma unroll
    for (int s = 0; s < 4; ++s) {
      int n = ng * 64 + s * 16 + lo;             // W row n == B column n
      v16h bf = load_bfrag(W, n, C_DIM, k0, lhi);
      acc[0][s] = wmma_f16(a0, bf, acc[0][s]);
      acc[1][s] = wmma_f16(a1, bf, acc[1][s]);
    }
  }

#pragma unroll
  for (int t = 0; t < 2; ++t) {
    int mt  = mt2 * 2 + t;                       // pair may straddle batches
    int b   = mt / QT_PER;
    int lq0 = (mt - b * QT_PER) * 16;
    int bh  = b * NH + ng;
#pragma unroll
    for (int s = 0; s < 4; ++s) {
      float bb = bias[ng * 64 + s * 16 + lo];
      int d = s * 16 + lo;
#pragma unroll
      for (int r = 0; r < 8; ++r) {              // D: lane holds col d, rows vary
        int lq = lq0 + lhi * 8 + r;
        _Float16 val = (_Float16)(acc[t][s][r] + bb);
        if (which == 0)      q16[((size_t)bh * LP + lq) * HD + d] = val;
        else if (which == 1) k16[((size_t)bh * LP + lq) * HD + d] = val;
        else                 vT16[((size_t)bh * HD + d) * LP + lq] = val;
      }
    }
  }
}

// Flash-style attention: one wave per (b, h, 16-row q tile).
// Wave-uniform skipping of fully-blocked 32-key tiles (mask is block-structured:
// 16-row q tiles and 32-key k tiles never straddle the 512/1536 region bounds).
__global__ void attn(const _Float16* __restrict__ q16,
                     const _Float16* __restrict__ k16,
                     const _Float16* __restrict__ vT16,
                     _Float16* __restrict__ y16, int nwork) {
  __shared__ __align__(16) _Float16 Pst[4 * 16 * 32];  // per-wave P tile
  int wslot = threadIdx.x >> 5;
  int wid = blockIdx.x * (blockDim.x >> 5) + wslot;
  if (wid >= nwork) return;
  _Float16* P = Pst + wslot * 512;
  int lane = threadIdx.x & 31;
  int lo = lane & 15, lhi = lane >> 4;
  int bh = wid / QT_PER;
  int qt = wid - bh * QT_PER;
  int b = bh / NH, h = bh - b * NH;
  const _Float16* qb = q16 + (size_t)bh * LP * HD;
  const _Float16* kb = k16 + (size_t)bh * LP * HD;
  const _Float16* vb = vT16 + (size_t)bh * HD * LP;

  // Q fragments (hd=64 -> two chained 16x32 A frags), 1/sqrt(64) folded in.
  union { v16h v; _Float16 e[16]; } uq0, uq1;
  uq0.v = load_afrag(qb, qt * 16 + lo, HD, 0, lhi);
  uq1.v = load_afrag(qb, qt * 16 + lo, HD, 32, lhi);
#pragma unroll
  for (int i = 0; i < 16; ++i) {
    uq0.e[i] = uq0.e[i] * (_Float16)0.125f;
    uq1.e[i] = uq1.e[i] * (_Float16)0.125f;
  }

  float mrun[8], ssum[8];
#pragma unroll
  for (int r = 0; r < 8; ++r) { mrun[r] = -1e30f; ssum[r] = 0.f; }
  v8f zero = {0.f,0.f,0.f,0.f,0.f,0.f,0.f,0.f};
  v8f o[4];
#pragma unroll
  for (int s = 0; s < 4; ++s) o[s] = zero;

  // wave-uniform q-tile classification
  bool qtext  = (qt * 16) >= ML;
  int  qreg   = (qt * 16) >> 9;                  // 0..2 when motion
  int  qq_max = qt * 16 + 15 - qreg * T_MOT;     // max in-region q offset
  int  qbase  = qt * 16 + lhi * 8;               // rows this lane-half owns
  int  qq_me  = qbase - qreg * T_MOT;            // in-region offset of row r=0

  for (int kt = 0; kt < LP / 32; ++kt) {
    int k0 = kt * 32;
    bool ktext = (k0 >= ML);
    int  kreg  = k0 >> 9;
    // ---- wave-uniform skip of fully blocked tiles ----
    bool process;
    if (qtext)       process = ktext;                         // text->text only
    else if (ktext)  process = (qreg > 0);                    // rows [T,3T)->text
    else if (qreg == 0 && kreg != 0) process = false;
    else             process = ((k0 - kreg * T_MOT) <= qq_max); // causal-ish
    if (!process) continue;

    // mask mode for this tile: 0 = allowed (pad check only), 1 = causal compare
    bool domask = (!qtext && !ktext);
    bool strict = (qreg == 1 && kreg != 0) || (qreg == 2 && kreg == 2);

    int kc0 = k0 + lo;
    int kc1 = k0 + 16 + lo;
    // scores: S = (Q * 0.125) @ K^T   (K row-major == B col-major layout)
    v8f s0 = wmma_f16(uq0.v, load_bfrag(kb, kc0, HD, 0, lhi), zero);
    s0     = wmma_f16(uq1.v, load_bfrag(kb, kc0, HD, 32, lhi), s0);
    v8f s1 = wmma_f16(uq0.v, load_bfrag(kb, kc1, HD, 0, lhi), zero);
    s1     = wmma_f16(uq1.v, load_bfrag(kb, kc1, HD, 32, lhi), s1);
    bool pad0 = (kc0 >= L_SEQ), pad1 = (kc1 >= L_SEQ);
    int  kk0 = kc0 - kreg * T_MOT, kk1 = kc1 - kreg * T_MOT;

    float sc[8];
#pragma unroll
    for (int r = 0; r < 8; ++r) {
      int qq = qq_me + r;
      bool blk0 = pad0, blk1 = pad1;
      if (domask) {
        blk0 = blk0 || (strict ? (qq <= kk0) : (qq < kk0));
        blk1 = blk1 || (strict ? (qq <= kk1) : (qq < kk1));
      }
      float a0 = blk0 ? -1e30f : s0[r];
      float a1 = blk1 ? -1e30f : s1[r];
      // row max across the 16 lanes of this half (xor masks stay in-group)
      float m = fmaxf(a0, a1);
      m = fmaxf(m, __shfl_xor(m, 1, 32));
      m = fmaxf(m, __shfl_xor(m, 2, 32));
      m = fmaxf(m, __shfl_xor(m, 4, 32));
      m = fmaxf(m, __shfl_xor(m, 8, 32));
      float mnew = fmaxf(mrun[r], m);
      sc[r] = __expf(mrun[r] - mnew);            // ==0 when real scores arrive
      mrun[r] = mnew;
      float p0 = __expf(a0 - mnew);
      float p1 = __expf(a1 - mnew);
      P[(lhi * 8 + r) * 32 + lo]      = (_Float16)p0;   // stage P for A-layout
      P[(lhi * 8 + r) * 32 + 16 + lo] = (_Float16)p1;
      float ps = p0 + p1;
      ps += __shfl_xor(ps, 1, 32);
      ps += __shfl_xor(ps, 2, 32);
      ps += __shfl_xor(ps, 4, 32);
      ps += __shfl_xor(ps, 8, 32);
      ssum[r] = ssum[r] * sc[r] + ps;
    }
    asm volatile("s_wait_dscnt 0x0" ::: "memory"); // P visible before re-read
    union { v16h v; v8h hh[2]; } pa;               // P as 16x32 A fragment
    const _Float16* pr = P + lo * 32 + lhi * 16;
    pa.hh[0] = *(const v8h*)pr;
    pa.hh[1] = *(const v8h*)(pr + 8);
#pragma unroll
    for (int s = 0; s < 4; ++s) {
#pragma unroll
      for (int r = 0; r < 8; ++r) o[s][r] = o[s][r] * sc[r];
      v16h bf = load_bfrag(vb, s * 16 + lo, LP, k0, lhi); // vT rows = B cols
      o[s] = wmma_f16(pa.v, bf, o[s]);
    }
  }

#pragma unroll
  for (int s = 0; s < 4; ++s) {
    int d = h * HD + s * 16 + lo;
#pragma unroll
    for (int r = 0; r < 8; ++r) {
      int gq = qbase + r;
      y16[((size_t)b * L_SEQ + gq) * C_DIM + d] = (_Float16)(o[s][r] / ssum[r]);
    }
  }
}

// Output projection: y16 [B*L x C] @ Wp^T + bp -> f32 out. 32x64 tile per wave.
__global__ void out_proj(const _Float16* __restrict__ y16,
                         const _Float16* __restrict__ Wp16,
                         const float* __restrict__ bp,
                         float* __restrict__ out, int nwork) {
  int wid = blockIdx.x * (blockDim.x >> 5) + (threadIdx.x >> 5);
  if (wid >= nwork) return;
  int lane = threadIdx.x & 31;
  int lo = lane & 15, lhi = lane >> 4;
  int ng  = wid % NH;
  int mt2 = wid / NH;

  v8f zero = {0.f,0.f,0.f,0.f,0.f,0.f,0.f,0.f};
  v8f acc[2][4];
#pragma unroll
  for (int t = 0; t < 2; ++t)
#pragma unroll
    for (int s = 0; s < 4; ++s) acc[t][s] = zero;

  int arow0 = mt2 * 32 + lo;
  for (int k0 = 0; k0 < C_DIM; k0 += 32) {
    v16h a0 = load_afrag(y16, arow0,      C_DIM, k0, lhi);
    v16h a1 = load_afrag(y16, arow0 + 16, C_DIM, k0, lhi);
#pragma unroll
    for (int s = 0; s < 4; ++s) {
      int n = ng * 64 + s * 16 + lo;
      v16h bf = load_bfrag(Wp16, n, C_DIM, k0, lhi);
      acc[0][s] = wmma_f16(a0, bf, acc[0][s]);
      acc[1][s] = wmma_f16(a1, bf, acc[1][s]);
    }
  }
#pragma unroll
  for (int t = 0; t < 2; ++t) {
#pragma unroll
    for (int s = 0; s < 4; ++s) {
      float bb = bp[ng * 64 + s * 16 + lo];
#pragma unroll
      for (int r = 0; r < 8; ++r) {
        out[((size_t)(mt2 * 32 + t * 16 + lhi * 8 + r)) * C_DIM +
            ng * 64 + s * 16 + lo] = acc[t][s][r] + bb;
      }
    }
  }
}

extern "C" void kernel_launch(void* const* d_in, const int* in_sizes, int n_in,
                              void* d_out, int out_size, void* d_ws, size_t ws_size,
                              hipStream_t stream) {
  const float* x  = (const float*)d_in[0];
  const float* Wq = (const float*)d_in[1];
  const float* bq = (const float*)d_in[2];
  const float* Wk = (const float*)d_in[3];
  const float* bk = (const float*)d_in[4];
  const float* Wv = (const float*)d_in[5];
  const float* bv = (const float*)d_in[6];
  const float* Wp = (const float*)d_in[7];
  const float* bp = (const float*)d_in[8];
  float* out = (float*)d_out;

  const int nx = in_sizes[0];                 // B*L*C
  const int B  = nx / (L_SEQ * C_DIM);        // 4
  const int nw = C_DIM * C_DIM;

  // workspace carve-up (all region sizes are multiples of 256 bytes)
  char* ws = (char*)d_ws;
  size_t off = 0;
  _Float16* x16  = (_Float16*)(ws + off); off += (size_t)nx * 2;
  _Float16* Wq16 = (_Float16*)(ws + off); off += (size_t)nw * 2;
  _Float16* Wk16 = (_Float16*)(ws + off); off += (size_t)nw * 2;
  _Float16* Wv16 = (_Float16*)(ws + off); off += (size_t)nw * 2;
  _Float16* Wp16 = (_Float16*)(ws + off); off += (size_t)nw * 2;
  size_t sq = (size_t)B * NH * LP * HD * 2;
  _Float16* q16  = (_Float16*)(ws + off); off += sq;
  _Float16* k16  = (_Float16*)(ws + off); off += sq;
  _Float16* vT16 = (_Float16*)(ws + off); off += sq;
  _Float16* y16  = (_Float16*)(ws + off); off += (size_t)nx * 2;

  const int thr = 256;
  cvt_f32_f16<<<(nx + thr - 1) / thr, thr, 0, stream>>>(x, x16, nx);
  cvt_f32_f16<<<(nw + thr - 1) / thr, thr, 0, stream>>>(Wq, Wq16, nw);
  cvt_f32_f16<<<(nw + thr - 1) / thr, thr, 0, stream>>>(Wk, Wk16, nw);
  cvt_f32_f16<<<(nw + thr - 1) / thr, thr, 0, stream>>>(Wv, Wv16, nw);
  cvt_f32_f16<<<(nw + thr - 1) / thr, thr, 0, stream>>>(Wp, Wp16, nw);

  int mpairs = (B * QT_PER) / 2;              // 202 (404 tiles, even)
  int nq = mpairs * NH * 3;                   // 7272 waves
  qkv_proj<<<(nq + 3) / 4, 128, 0, stream>>>(x16, Wq16, bq, Wk16, bk, Wv16, bv,
                                             q16, k16, vT16, nq);
  int na = B * NH * QT_PER;                   // 4848 waves
  attn<<<(na + 3) / 4, 128, 0, stream>>>(q16, k16, vT16, y16, na);
  int np = mpairs * NH;                       // 2424 waves
  out_proj<<<(np + 3) / 4, 128, 0, stream>>>(y16, Wp16, bp, out, np);
}
